// CAM_58428735094866
// MI455X (gfx1250) — matrix operand order repaired and checked by reading, not compiled, not run on hardware
//
#include <hip/hip_runtime.h>

// ---------------------------------------------------------------------------
// CAM (channel attention) for x:[2,48,48,48,256] fp32.
//   aTa = A^T A  (bf16 WMMA, f32 accum, split-K atomics)
//   S   = softmax_rows(aTa)  (fp32, stored transposed as bf16 St[j][c])
//   out = gamma * (A @ S) + x  (bf16 WMMA, f32 accum, fused epilogue)
// Memory-bound target: ~680 MB of HBM traffic @ 23.3 TB/s.
// ---------------------------------------------------------------------------

typedef __attribute__((ext_vector_type(16))) __bf16 bf16x16;
typedef __attribute__((ext_vector_type(8)))  __bf16 bf16x8;
typedef __attribute__((ext_vector_type(4)))  __bf16 bf16x4;
typedef __attribute__((ext_vector_type(2)))  __bf16 bf16x2;
typedef __attribute__((ext_vector_type(8)))  float  v8f;
typedef __attribute__((ext_vector_type(4)))  unsigned int u32x4;
typedef __attribute__((ext_vector_type(4)))  int    i32x4;
typedef __attribute__((ext_vector_type(8)))  int    i32x8;

#define CAM_B      2
#define CAM_C      256
#define CAM_L      110592     // 48^3
#define P1_CHUNKS  128        // split-K chunks per batch
#define P1_CHUNK_L 864        // 110592 / 128
#define P1_KSTEPS  27         // 864 / 32
#define P1_PITCH   40         // padded k-pitch (elems): even + multiple of 8
#define P3_WGS     216        // 6912 M-tiles / 32 per WG

__device__ __forceinline__ __bf16 f2bf(float f) {
  unsigned u = __builtin_bit_cast(unsigned, f);
  unsigned r = u + 0x7fffu + ((u >> 16) & 1u);   // round-to-nearest-even
  unsigned short h = (unsigned short)(r >> 16);
  return __builtin_bit_cast(__bf16, h);
}

// Store two floats as a packed bf16 pair (one 32-bit DS store).
__device__ __forceinline__ void st_bf16_pair(__bf16* d, float lo, float hi) {
  unsigned p = ((unsigned)__builtin_bit_cast(unsigned short, f2bf(hi)) << 16) |
                (unsigned)__builtin_bit_cast(unsigned short, f2bf(lo));
  *(unsigned*)(void*)d = p;
}

// Build a 16-wide bf16 A-fragment from 16 fp32 register values.
__device__ __forceinline__ bf16x16 mk_bf16x16(float4 f0, float4 f1,
                                              float4 f2, float4 f3) {
  bf16x16 r = { f2bf(f0.x), f2bf(f0.y), f2bf(f0.z), f2bf(f0.w),
                f2bf(f1.x), f2bf(f1.y), f2bf(f1.z), f2bf(f1.w),
                f2bf(f2.x), f2bf(f2.y), f2bf(f2.z), f2bf(f2.w),
                f2bf(f3.x), f2bf(f3.y), f2bf(f3.z), f2bf(f3.w) };
  return r;
}

// ---------------------------------------------------------------------------
// Phase 1: partial Gram aTa[b,i,j] += sum over this WG's L-chunk.
// WG = 16 waves. LDS slab lt[c][k] (bf16, transposed) so A and B fragments
// are contiguous 16B ds_load_b128's. Staging handles k-PAIRS so conversion
// packs into single b32 DS stores (merged into ds_store_2addr_b32).
// B fragments are software-pipelined: loads for fragment j+1 issue before
// the WMMA consuming fragment j, so only partial s_wait_dscnt is needed.
// ---------------------------------------------------------------------------
__global__ void __launch_bounds__(512) cam_gram(const float* __restrict__ x,
                                                float* __restrict__ aTa)
{
  __shared__ __bf16 lt[CAM_C * P1_PITCH];        // 20480 B
  const int b    = blockIdx.y;
  const int chnk = blockIdx.x;
  const int tid  = (int)threadIdx.x;
  const int wave = tid >> 5;
  const int lane = tid & 31;
  const int m    = lane & 15;                    // row-in-tile / column lane
  const int kh   = lane >> 4;                    // K half selector
  const int i0   = wave << 4;                    // this wave's i-tile base
  const float* xb = x + (long long)b * CAM_L * CAM_C;

  const v8f vz = {0.f,0.f,0.f,0.f,0.f,0.f,0.f,0.f};
  v8f acc[16];
#pragma unroll
  for (int jt = 0; jt < 16; ++jt) acc[jt] = vz;

  // staging map: wave = k-pair index (2 rows of 32), lane covers 8 channels
  const int kp = wave;                           // k-pair 0..15 -> k = 2kp,2kp+1
  const int c0 = lane * 8;                       // channel base 0..248

  auto ldb = [&](int jt) -> bf16x16 {
    const __bf16* bp = &lt[(jt * 16 + m) * P1_PITCH + kh * 8];
    return __builtin_shufflevector(*(const bf16x8*)bp, *(const bf16x8*)(bp + 16),
        0,1,2,3,4,5,6,7,8,9,10,11,12,13,14,15);
  };

  for (int ks = 0; ks < P1_KSTEPS; ++ks) {
    const long long l0 = (long long)chnk * P1_CHUNK_L + (long long)ks * 32;
    __syncthreads();
    {
      const float* r0 = xb + (l0 + 2 * kp) * CAM_C + c0;
      const float* r1 = r0 + CAM_C;
      if (ks + 1 < P1_KSTEPS)                    // prefetch next slab
        __builtin_prefetch((const void*)(r0 + 32 * CAM_C), 0, 0);
      float4 e0 = *(const float4*)(r0);
      float4 e1 = *(const float4*)(r0 + 4);
      float4 o0 = *(const float4*)(r1);
      float4 o1 = *(const float4*)(r1 + 4);
      const float ev[8] = { e0.x, e0.y, e0.z, e0.w, e1.x, e1.y, e1.z, e1.w };
      const float ov[8] = { o0.x, o0.y, o0.z, o0.w, o1.x, o1.y, o1.z, o1.w };
#pragma unroll
      for (int c = 0; c < 8; ++c)
        st_bf16_pair(&lt[(c0 + c) * P1_PITCH + 2 * kp], ev[c], ov[c]);
    }
    __syncthreads();

    // A fragment (16x32 bf16): lane holds M=m, K in {kh*8..+7} U {kh*8+16..+23}
    bf16x8 a0 = *(const bf16x8*)&lt[(i0 + m) * P1_PITCH + kh * 8];
    bf16x8 a1 = *(const bf16x8*)&lt[(i0 + m) * P1_PITCH + kh * 8 + 16];
    bf16x16 afrag = __builtin_shufflevector(a0, a1,
        0,1,2,3,4,5,6,7,8,9,10,11,12,13,14,15);

    // software-pipelined B fragments
    bf16x16 bcur = ldb(0);
#pragma unroll
    for (int jt = 0; jt < 16; ++jt) {
      bf16x16 bnxt = bcur;
      if (jt < 15) bnxt = ldb(jt + 1);
      acc[jt] = __builtin_amdgcn_wmma_f32_16x16x32_bf16(
          false, afrag, false, bcur, (short)0, acc[jt], false, false);
      bcur = bnxt;
    }
  }

  // C/D layout: VGPR r -> M = r + 8*kh, N = m (lane&15)
  float* dst = aTa + ((long long)b * CAM_C + i0) * CAM_C;
#pragma unroll
  for (int jt = 0; jt < 16; ++jt)
#pragma unroll
    for (int r = 0; r < 8; ++r)
      atomicAdd(&dst[(r + 8 * kh) * CAM_C + jt * 16 + m], acc[jt][r]);
}

// ---------------------------------------------------------------------------
// Phase 2: row softmax of aTa; write transposed bf16 St[b][j][c] = S[b][c][j].
// ---------------------------------------------------------------------------
__global__ void __launch_bounds__(256) cam_softmax(const float* __restrict__ aTa,
                                                   __bf16* __restrict__ St)
{
  __shared__ float red[256];
  const int row = (int)blockIdx.x;               // b*256 + i
  const int b = row >> 8;
  const int i = row & 255;
  const int t = (int)threadIdx.x;
  const float v = aTa[(long long)row * 256 + t];
  red[t] = v;
  __syncthreads();
  for (int s = 128; s > 0; s >>= 1) {
    if (t < s) red[t] = fmaxf(red[t], red[t + s]);
    __syncthreads();
  }
  const float mx = red[0];
  __syncthreads();
  const float e = __expf(v - mx);
  red[t] = e;
  __syncthreads();
  for (int s = 128; s > 0; s >>= 1) {
    if (t < s) red[t] += red[t + s];
    __syncthreads();
  }
  const float inv = 1.0f / red[0];
  St[(long long)b * (CAM_C * CAM_C) + (long long)t * CAM_C + i] = f2bf(e * inv);
}

// ---------------------------------------------------------------------------
// Phase 3: out = gamma * (A @ S) + x. St for this batch (128 KB bf16) staged
// into LDS once per WG (TDM tensor_load_to_lds when available), then each
// wave streams 16x16 M-tiles: A from global (contiguous K), B from LDS.
// ---------------------------------------------------------------------------
#ifndef CAM_USE_TDM
#if __has_builtin(__builtin_amdgcn_tensor_load_to_lds)
#define CAM_USE_TDM 1
#else
#define CAM_USE_TDM 0
#endif
#endif

__global__ void __launch_bounds__(512) cam_av(const float* __restrict__ x,
                                              const __bf16* __restrict__ St,
                                              const float* __restrict__ gamma,
                                              float* __restrict__ out)
{
  extern __shared__ __bf16 sS[];                 // 65536 bf16 = 128 KB (St[j][c])
  const int b    = blockIdx.y;
  const int bx   = blockIdx.x;
  const int tid  = (int)threadIdx.x;
  const int wave = tid >> 5;
  const int lane = tid & 31;
  const int m    = lane & 15;
  const int kh   = lane >> 4;
  const __bf16* Sb = St + (long long)b * (CAM_C * CAM_C);

#if CAM_USE_TDM
  if (wave == 0) {
    // Tensor DMA descriptor (D#): 2D tile 256x256, 2-byte elems, global->LDS.
    unsigned long long ga = (unsigned long long)(const void*)Sb;
    unsigned ldsOff = (unsigned)(unsigned long long)(void*)&sS[0];
    u32x4 g0 = { 1u,                                      // count=1
                 ldsOff,                                  // lds_addr
                 (unsigned)(ga & 0xffffffffu),            // global_addr[31:0]
                 (unsigned)((ga >> 32) & 0x01ffffffu) | (2u << 30) }; // [56:32]|type=2
    i32x8 g1 = { (int)0x00010000,   // wg_mask=0, data_size=1 (2B)
                 (int)0x01000000,   // tensor_dim0 = 256 (bits 79:48)
                 (int)0x01000000,   // tensor_dim1 = 256 (bits 111:80)
                 (int)0x01000000,   // tile_dim0   = 256 (bits 127:112)
                 (int)0x00000100,   // tile_dim1   = 256, tile_dim2 = 0
                 (int)0x00000100,   // tensor_dim0_stride = 256
                 0, 0 };            // tensor_dim1_stride = 0
    i32x4 g2 = {0,0,0,0};
    i32x4 g3 = {0,0,0,0};
#if __clang_major__ >= 23
    i32x8 g4 = {0,0,0,0,0,0,0,0};
    __builtin_amdgcn_tensor_load_to_lds(g0, g1, g2, g3, g4, 0);
#else
    __builtin_amdgcn_tensor_load_to_lds(g0, g1, g2, g3, 0);
#endif
#if __has_builtin(__builtin_amdgcn_s_wait_tensorcnt)
    __builtin_amdgcn_s_wait_tensorcnt(0);
#endif
  }
  __syncthreads();
#else
  {
    const uint4* src = (const uint4*)Sb;
    uint4* dst = (uint4*)&sS[0];
    for (int q = tid; q < 8192; q += 512) dst[q] = src[q];
  }
  __syncthreads();
#endif

  const float g = gamma[0];
  const v8f vz = {0.f,0.f,0.f,0.f,0.f,0.f,0.f,0.f};

  auto ldb = [&](int jt, int kk) -> bf16x16 {
    const __bf16* bp = &sS[(jt * 16 + m) * CAM_C + kk * 32 + kh * 8];
    return __builtin_shufflevector(*(const bf16x8*)bp, *(const bf16x8*)(bp + 16),
        0,1,2,3,4,5,6,7,8,9,10,11,12,13,14,15);
  };

#pragma unroll 1
  for (int pass = 0; pass < 2; ++pass) {
    const int mt = bx * 32 + pass * 16 + wave;   // M-tile index, < 6912
    const long long l0 = (long long)mt * 16;
    const float* xrow = x + ((long long)b * CAM_L + l0 + m) * CAM_C;
    v8f acc[16];
#pragma unroll
    for (int jt = 0; jt < 16; ++jt) acc[jt] = vz;

#pragma unroll 1
    for (int kk = 0; kk < 8; ++kk) {
      const int kb = kk * 32 + kh * 8;
      float4 f0 = *(const float4*)(xrow + kb);
      float4 f1 = *(const float4*)(xrow + kb + 4);
      float4 f2 = *(const float4*)(xrow + kb + 16);
      float4 f3 = *(const float4*)(xrow + kb + 20);
      bf16x16 afrag = mk_bf16x16(f0, f1, f2, f3);

      // software-pipelined B fragments
      bf16x16 bcur = ldb(0, kk);
#pragma unroll
      for (int jt = 0; jt < 16; ++jt) {
        bf16x16 bnxt = bcur;
        if (jt < 15) bnxt = ldb(jt + 1, kk);
        acc[jt] = __builtin_amdgcn_wmma_f32_16x16x32_bf16(
            false, afrag, false, bcur, (short)0, acc[jt], false, false);
        bcur = bnxt;
      }
    }

    const long long obase = ((long long)b * CAM_L + l0) * CAM_C;
#pragma unroll
    for (int jt = 0; jt < 16; ++jt)
#pragma unroll
      for (int r = 0; r < 8; ++r) {
        const long long idx = obase + (long long)(r + 8 * kh) * CAM_C + jt * 16 + m;
        out[idx] = g * acc[jt][r] + x[idx];
      }
  }
}

// ---------------------------------------------------------------------------
__global__ void cam_zero(float* __restrict__ p, int n) {
  int i = blockIdx.x * blockDim.x + threadIdx.x;
  if (i < n) p[i] = 0.0f;
}

extern "C" void kernel_launch(void* const* d_in, const int* in_sizes, int n_in,
                              void* d_out, int out_size, void* d_ws, size_t ws_size,
                              hipStream_t stream) {
  (void)in_sizes; (void)n_in; (void)out_size; (void)ws_size;
  const float* x     = (const float*)d_in[0];
  const float* gamma = (const float*)d_in[1];
  float* out = (float*)d_out;

  float*  aTa = (float*)d_ws;                                        // 512 KB
  __bf16* St  = (__bf16*)((char*)d_ws +
                 (size_t)CAM_B * CAM_C * CAM_C * sizeof(float));     // 256 KB

  const int nATA = CAM_B * CAM_C * CAM_C;
  cam_zero<<<(nATA + 255) / 256, 256, 0, stream>>>(aTa, nATA);
  cam_gram<<<dim3(P1_CHUNKS, CAM_B), 512, 0, stream>>>(x, aTa);
  cam_softmax<<<CAM_B * CAM_C, 256, 0, stream>>>(aTa, St);

  (void)hipFuncSetAttribute((const void*)cam_av,
                            hipFuncAttributeMaxDynamicSharedMemorySize,
                            CAM_C * CAM_C * (int)sizeof(__bf16));    // 128 KB
  cam_av<<<dim3(P3_WGS, CAM_B), 512, CAM_C * CAM_C * sizeof(__bf16), stream>>>(
      x, St, gamma, out);
}
